// Box2FeatureGeneratorV2_29901562315078
// MI455X (gfx1250) — matrix operand order repaired and checked by reading, not compiled, not run on hardware
//
#include <hip/hip_runtime.h>

typedef __attribute__((ext_vector_type(16))) __bf16 v16bf;
typedef __attribute__((ext_vector_type(8)))  float  v8f;

#define H_GRID   200
#define W_GRID   704
#define NPIX     (H_GRID * W_GRID)
#define C_CH     256
#define N_BOX    128
#define WSTRIPS  44                  // 704 / 16
#define NSTRIP   (H_GRID * WSTRIPS)  // 8800 pixel strips
#define NWAVE2   (NSTRIP * 2)        // 2 waves per strip (128 channels each)
#define KSTEPS_CONV 72               // 9*256 / 32
#define KSTEPS_SC   4                // 128 / 32
#define CB_PER_WAVE 8                // 8 x 16 = 128 channels per wave

// ---------------------------------------------------------------- helpers

__device__ inline unsigned pack_bf2(float a, float b) {
    union { __bf16 h; unsigned short u; } ua, ub;
    ua.h = (__bf16)a; ub.h = (__bf16)b;
    return (unsigned)ua.u | ((unsigned)ub.u << 16);
}

__device__ inline float bf_to_f(unsigned short u) {
    union { __bf16 h; unsigned short s; } c; c.s = u; return (float)c.h;
}

union BF16x16 { uint4 q[2]; v16bf v; };

__device__ inline v16bf load_bf16x16(const __bf16* p) {
    BF16x16 u;
    const uint4* q = (const uint4*)p;
    u.q[0] = q[0]; u.q[1] = q[1];
    return u.v;
}

// ---------------------------------------------------------------- MLP phase

__global__ void feat_kernel(const float* __restrict__ box,
                            const float* __restrict__ score,
                            float* __restrict__ feat) {
    int i = blockIdx.x * blockDim.x + threadIdx.x;
    if (i >= N_BOX * 25) return;
    int n = i / 25, j = i % 25;
    feat[i] = (j < 24) ? box[n * 24 + j] : score[n];
}

// out[n][c] = act(sum_k in[n][k]*w[k][c] + b[c]); mode 1: relu, mode 2: *score
__global__ void mlp_kernel(const float* __restrict__ in,
                           const float* __restrict__ w,
                           const float* __restrict__ bias,
                           float* __restrict__ out,
                           int N, int K, int Cout,
                           const float* __restrict__ score, int mode) {
    int idx = blockIdx.x * blockDim.x + threadIdx.x;
    if (idx >= N * Cout) return;
    int n = idx / Cout, c = idx % Cout;
    float s = bias[c];
    for (int k = 0; k < K; ++k) s = fmaf(in[n * K + k], w[k * Cout + c], s);
    if (mode == 1) s = fmaxf(s, 0.f);
    if (mode == 2) s *= score[n];
    out[idx] = s;
}

// ---------------------------------------------------------------- prep phase

// Edge half-plane params: inside iff eA*cx + eB*cy + eC >= 0 for all 4 edges.
__global__ void edge_kernel(const float* __restrict__ box,
                            float* __restrict__ eA,
                            float* __restrict__ eB,
                            float* __restrict__ eC) {
    int n = blockIdx.x * blockDim.x + threadIdx.x;
    if (n >= N_BOX) return;
    float gx[4], gy[4];
    for (int e = 0; e < 4; ++e) {
        gx[e] = (box[(n * 8 + e) * 3 + 0] - (-140.8f)) / 0.4f;
        gy[e] = (box[(n * 8 + e) * 3 + 1] - (-40.0f)) / 0.4f;
    }
    for (int e = 0; e < 4; ++e) {
        float ax = gx[e], ay = gy[e];
        float bx = gx[(e + 1) & 3], by = gy[(e + 1) & 3];
        float vx = bx - ax, vy = by - ay;
        eA[e * N_BOX + n] = -vy;
        eB[e * N_BOX + n] = vx;
        eC[e * N_BOX + n] = vy * ax - vx * ay;
    }
}

// Fused BN: scale = g/sqrt(v+eps); shift = b - m*scale.  idx over 6*256.
__global__ void bn_kernel(const float* __restrict__ g, const float* __restrict__ b,
                          const float* __restrict__ m, const float* __restrict__ v,
                          float* __restrict__ scale, float* __restrict__ shift) {
    int i = blockIdx.x * blockDim.x + threadIdx.x;
    if (i >= 6 * C_CH) return;
    float inv = g[i] * rsqrtf(v[i] + 1e-5f);
    scale[i] = inv;
    shift[i] = b[i] - m[i] * inv;
}

// Pack obj^T (A matrix, M=channels x K=boxes) into WMMA lane order.
// A lane layout (16-bit A 16x32): lane = m + 16*khi; element j:
//   k = khi*8 + (j&7) + ((j&8) ? 16 : 0)
__global__ void objpack_kernel(const float* __restrict__ obj, __bf16* __restrict__ ap) {
    int idx = blockIdx.x * blockDim.x + threadIdx.x;
    if (idx >= 16 * KSTEPS_SC * 32 * 16) return;
    int j     = idx & 15;
    int lane  = (idx >> 4) & 31;
    int kstep = (idx >> 9) & 3;
    int coBlk = idx >> 11;
    int m = lane & 15, khi = lane >> 4;
    int k  = kstep * 32 + khi * 8 + (j & 7) + ((j & 8) ? 16 : 0);
    int co = coBlk * 16 + m;
    ap[idx] = (__bf16)obj[k * C_CH + co];
}

// Pack conv weights for all 6 convs into WMMA A lane order.
// K ordering is tap-major: k = tap*256 + ci (tap = 3*(dr+1)+(dc+1)).
__global__ void wpack_kernel(const float* __restrict__ w, __bf16* __restrict__ wp,
                             long total) {
    long idx = (long)blockIdx.x * blockDim.x + threadIdx.x;
    if (idx >= total) return;
    long t = idx;
    int j = (int)(t & 15); t >>= 4;
    int lane = (int)(t & 31); t >>= 5;
    int kstep = (int)(t % KSTEPS_CONV); t /= KSTEPS_CONV;
    int coBlk = (int)(t & 15); t >>= 4;
    int convId = (int)t;              // 0..5  (blk*2 + layer)
    int m = lane & 15, khi = lane >> 4;
    int k = kstep * 32 + khi * 16 + j;
    int tap = k >> 8, ci = k & 255;
    int co = coBlk * 16 + m;
    long widx = ((((long)convId * C_CH + co) * C_CH + ci) * 3 + (tap / 3)) * 3 + (tap % 3);
    wp[idx] = (__bf16)w[widx];
}

// ------------------------------------------------------- raster+scatter (WMMA)

// One wave = 16-pixel strip x 128 channels (2 waves per strip).
// C[c, pix] = sum_n obj[n,c] * inside(n, pix); divide by count; store bf16 HWC.
__global__ void __launch_bounds__(256, 1)
scatter_kernel(const float* __restrict__ eA,
               const float* __restrict__ eB,
               const float* __restrict__ eC,
               const __bf16* __restrict__ objApack,
               __bf16* __restrict__ actX) {
    int tid  = blockIdx.x * blockDim.x + threadIdx.x;
    int wave = tid >> 5;
    int lane = tid & 31;
    if (wave >= NWAVE2) return;
    int half  = wave & 1;           // which 128-channel half
    int strip = wave >> 1;
    int y  = strip / WSTRIPS;
    int xs = (strip % WSTRIPS) * 16;
    int p = lane & 15, khi = lane >> 4;
    int px = xs + p;
    float cx = (float)px + 0.5f;
    float cy = (float)y  + 0.5f;

    v8f acc[CB_PER_WAVE];
#pragma unroll
    for (int cb = 0; cb < CB_PER_WAVE; ++cb)
#pragma unroll
        for (int r = 0; r < 8; ++r) acc[cb][r] = 0.f;

    float cnt = 0.f;
    for (int ks = 0; ks < KSTEPS_SC; ++ks) {
        // Build B (K=32 boxes x N=16 pixels): lane holds K = khi*8 + (j&7) + ((j&8)?16:0)
        v16bf bv;
#pragma unroll
        for (int j = 0; j < 16; ++j) {
            int n = ks * 32 + khi * 8 + (j & 7) + ((j & 8) ? 16 : 0);
            float s0 = fmaf(eA[0 * N_BOX + n], cx, fmaf(eB[0 * N_BOX + n], cy, eC[0 * N_BOX + n]));
            float s1 = fmaf(eA[1 * N_BOX + n], cx, fmaf(eB[1 * N_BOX + n], cy, eC[1 * N_BOX + n]));
            float s2 = fmaf(eA[2 * N_BOX + n], cx, fmaf(eB[2 * N_BOX + n], cy, eC[2 * N_BOX + n]));
            float s3 = fmaf(eA[3 * N_BOX + n], cx, fmaf(eB[3 * N_BOX + n], cy, eC[3 * N_BOX + n]));
            float f = (s0 >= 0.f && s1 >= 0.f && s2 >= 0.f && s3 >= 0.f) ? 1.f : 0.f;
            cnt += f;
            bv[j] = (__bf16)f;
        }
#pragma unroll
        for (int cb = 0; cb < CB_PER_WAVE; ++cb) {
            int cbGlob = half * CB_PER_WAVE + cb;
            v16bf av = load_bf16x16(objApack + (((size_t)cbGlob * KSTEPS_SC + ks) * 32 + lane) * 16);
            acc[cb] = __builtin_amdgcn_wmma_f32_16x16x32_bf16(
                false, av, false, bv, (short)0, acc[cb], false, false);
        }
    }

    // lanes p and p+16 covered the two K-halves of the same pixel
    cnt += __shfl_xor(cnt, 16, 32);
    float inv = 1.f / fmaxf(cnt, 1.f);

    size_t pixOff = ((size_t)y * W_GRID + px) * C_CH;
#pragma unroll
    for (int cb = 0; cb < CB_PER_WAVE; ++cb) {
        int cbase = (half * CB_PER_WAVE + cb) * 16 + khi * 8;
        uint4 o;
        o.x = pack_bf2(acc[cb][0] * inv, acc[cb][1] * inv);
        o.y = pack_bf2(acc[cb][2] * inv, acc[cb][3] * inv);
        o.z = pack_bf2(acc[cb][4] * inv, acc[cb][5] * inv);
        o.w = pack_bf2(acc[cb][6] * inv, acc[cb][7] * inv);
        *(uint4*)(actX + pixOff + cbase) = o;
    }
}

// ------------------------------------------------------- conv (implicit GEMM)

// One wave = 16-pixel strip x 128 out-channels (2 waves per strip).
// K = 9 taps * 256 ci, bf16 WMMA, f32 accumulate.
// Epilogue: BN scale/shift [+res] [relu]; store bf16 HWC; optional f32 CHW.
__global__ void __launch_bounds__(256, 1)
conv_kernel(const __bf16* __restrict__ in,
            const __bf16* __restrict__ wpack,
            const float* __restrict__ scale,
            const float* __restrict__ shift,
            const __bf16* __restrict__ res,
            __bf16* __restrict__ out,
            float* __restrict__ finalOut,
            int relu) {
    int tid  = blockIdx.x * blockDim.x + threadIdx.x;
    int wave = tid >> 5;
    int lane = tid & 31;
    if (wave >= NWAVE2) return;
    int half  = wave & 1;
    int strip = wave >> 1;
    int y  = strip / WSTRIPS;
    int xs = (strip % WSTRIPS) * 16;
    int p = lane & 15, khi = lane >> 4;
    int px = xs + p;

    v8f acc[CB_PER_WAVE];
#pragma unroll
    for (int cb = 0; cb < CB_PER_WAVE; ++cb)
#pragma unroll
        for (int r = 0; r < 8; ++r) acc[cb][r] = 0.f;

    const __bf16* wbase = wpack + (size_t)half * CB_PER_WAVE * KSTEPS_CONV * 32 * 16
                                + (size_t)lane * 16;

    for (int ks = 0; ks < KSTEPS_CONV; ++ks) {
        int tap = ks >> 3;
        int dr = tap / 3 - 1, dc = tap % 3 - 1;
        int yy = y + dr, xx = px + dc;
        // B: lane holds 16 contiguous ci at [yy][xx][ci0 .. ci0+15]
        BF16x16 u;
        if (yy >= 0 && yy < H_GRID && xx >= 0 && xx < W_GRID) {
            int ci0 = ((ks & 7) << 5) + khi * 16;
            const uint4* src = (const uint4*)(in + (((size_t)yy * W_GRID + xx) << 8) + ci0);
            u.q[0] = src[0]; u.q[1] = src[1];
        } else {
            u.q[0] = make_uint4(0u, 0u, 0u, 0u);
            u.q[1] = make_uint4(0u, 0u, 0u, 0u);
        }
        v16bf bv = u.v;
#pragma unroll
        for (int cb = 0; cb < CB_PER_WAVE; ++cb) {
            v16bf av = load_bf16x16(wbase + ((size_t)cb * KSTEPS_CONV + ks) * 32 * 16);
            acc[cb] = __builtin_amdgcn_wmma_f32_16x16x32_bf16(
                false, av, false, bv, (short)0, acc[cb], false, false);
        }
    }

    size_t pixOff = ((size_t)y * W_GRID + px) * C_CH;
#pragma unroll
    for (int cb = 0; cb < CB_PER_WAVE; ++cb) {
        int cbase = (half * CB_PER_WAVE + cb) * 16 + khi * 8;
        float vals[8];
#pragma unroll
        for (int r = 0; r < 8; ++r)
            vals[r] = fmaf(acc[cb][r], scale[cbase + r], shift[cbase + r]);
        if (res) {
            uint4 rq = *(const uint4*)(res + pixOff + cbase);
            unsigned ru[4] = { rq.x, rq.y, rq.z, rq.w };
#pragma unroll
            for (int r = 0; r < 8; ++r)
                vals[r] += bf_to_f((unsigned short)((r & 1) ? (ru[r >> 1] >> 16)
                                                            : (ru[r >> 1] & 0xffffu)));
        }
        if (relu) {
#pragma unroll
            for (int r = 0; r < 8; ++r) vals[r] = fmaxf(vals[r], 0.f);
        }
        uint4 o;
        o.x = pack_bf2(vals[0], vals[1]);
        o.y = pack_bf2(vals[2], vals[3]);
        o.z = pack_bf2(vals[4], vals[5]);
        o.w = pack_bf2(vals[6], vals[7]);
        *(uint4*)(out + pixOff + cbase) = o;
        if (finalOut) {
#pragma unroll
            for (int r = 0; r < 8; ++r)
                finalOut[(size_t)(cbase + r) * NPIX + (size_t)y * W_GRID + px] = vals[r];
        }
    }
}

// ---------------------------------------------------------------- launcher

extern "C" void kernel_launch(void* const* d_in, const int* in_sizes, int n_in,
                              void* d_out, int out_size, void* d_ws, size_t ws_size,
                              hipStream_t stream) {
    const float* pred_box   = (const float*)d_in[0];
    const float* pred_score = (const float*)d_in[1];
    const float* w1 = (const float*)d_in[2];
    const float* b1 = (const float*)d_in[3];
    const float* w2 = (const float*)d_in[4];
    const float* b2 = (const float*)d_in[5];
    const float* w3 = (const float*)d_in[6];
    const float* b3 = (const float*)d_in[7];
    const float* conv_w   = (const float*)d_in[8];
    const float* bn_gamma = (const float*)d_in[9];
    const float* bn_beta  = (const float*)d_in[10];
    const float* bn_mean  = (const float*)d_in[11];
    const float* bn_var   = (const float*)d_in[12];

    char* ws = (char*)d_ws;
    size_t off = 0;
    auto walloc = [&](size_t bytes) -> void* {
        void* ptr = ws + off;
        off = (off + bytes + 255) & ~(size_t)255;
        return ptr;
    };

    float* feat = (float*)walloc((size_t)N_BOX * 25 * 4);
    float* h1   = (float*)walloc((size_t)N_BOX * C_CH * 4);
    float* h2   = (float*)walloc((size_t)N_BOX * C_CH * 4);
    float* obj  = (float*)walloc((size_t)N_BOX * C_CH * 4);
    float* eA   = (float*)walloc((size_t)4 * N_BOX * 4);
    float* eB   = (float*)walloc((size_t)4 * N_BOX * 4);
    float* eC   = (float*)walloc((size_t)4 * N_BOX * 4);
    float* bnscale = (float*)walloc((size_t)6 * C_CH * 4);
    float* bnshift = (float*)walloc((size_t)6 * C_CH * 4);
    __bf16* objApack = (__bf16*)walloc((size_t)16 * KSTEPS_SC * 32 * 16 * 2);
    const size_t wpackPerConv = (size_t)16 * KSTEPS_CONV * 32 * 16;   // elements
    __bf16* wpack = (__bf16*)walloc(6 * wpackPerConv * 2);
    __bf16* actX  = (__bf16*)walloc((size_t)NPIX * C_CH * 2);
    __bf16* actH  = (__bf16*)walloc((size_t)NPIX * C_CH * 2);
    float*  outF  = (float*)d_out;

    // ---- MLP ----
    feat_kernel<<<(N_BOX * 25 + 255) / 256, 256, 0, stream>>>(pred_box, pred_score, feat);
    mlp_kernel<<<(N_BOX * C_CH + 255) / 256, 256, 0, stream>>>(
        feat, w1, b1, h1, N_BOX, 25, C_CH, nullptr, 1);
    mlp_kernel<<<(N_BOX * C_CH + 255) / 256, 256, 0, stream>>>(
        h1, w2, b2, h2, N_BOX, C_CH, C_CH, nullptr, 1);
    mlp_kernel<<<(N_BOX * C_CH + 255) / 256, 256, 0, stream>>>(
        h2, w3, b3, obj, N_BOX, C_CH, C_CH, pred_score, 2);

    // ---- prep ----
    edge_kernel<<<1, 128, 0, stream>>>(pred_box, eA, eB, eC);
    bn_kernel<<<(6 * C_CH + 255) / 256, 256, 0, stream>>>(
        bn_gamma, bn_beta, bn_mean, bn_var, bnscale, bnshift);
    objpack_kernel<<<(16 * KSTEPS_SC * 32 * 16 + 255) / 256, 256, 0, stream>>>(obj, objApack);
    {
        long total = 6L * wpackPerConv;
        wpack_kernel<<<(unsigned)((total + 255) / 256), 256, 0, stream>>>(conv_w, wpack, total);
    }

    // ---- raster + scatter GEMM (WMMA) ----
    scatter_kernel<<<(NWAVE2 * 32) / 256, 256, 0, stream>>>(eA, eB, eC, objApack, actX);

    // ---- 3 residual blocks, 2 convs each (WMMA implicit GEMM) ----
    dim3 cgrid((NWAVE2 * 32) / 256), cblk(256);
    for (int blk = 0; blk < 3; ++blk) {
        int cA = blk * 2 + 0, cB = blk * 2 + 1;
        // y = relu(bn(conv(x)))
        conv_kernel<<<cgrid, cblk, 0, stream>>>(
            actX, wpack + (size_t)cA * wpackPerConv,
            bnscale + (size_t)cA * C_CH, bnshift + (size_t)cA * C_CH,
            nullptr, actH, nullptr, 1);
        // x = relu(bn(conv(y)) + x)   (last block also writes f32 CHW output)
        conv_kernel<<<cgrid, cblk, 0, stream>>>(
            actH, wpack + (size_t)cB * wpackPerConv,
            bnscale + (size_t)cB * C_CH, bnshift + (size_t)cB * C_CH,
            actX, actX, (blk == 2) ? outF : nullptr, 1);
    }
}
